// Quantum_S4_14285061226522
// MI455X (gfx1250) — matrix operand order repaired and checked by reading, not compiled, not run on hardware
//
#include <hip/hip_runtime.h>

// ---------------------------------------------------------------------------
// Quantum S4 wavefunction forward pass for MI455X (gfx1250, wave32, WMMA).
//   amp, phase = f(spins) ; shapes (L,B,2) each, L = NSPIN+1.
// GLU linear (6.9e10 FLOP) runs on v_wmma_f32_16x16x32_bf16 with 8 output
// tiles per wave (A-fragment reuse) and W_glu fragments double-buffered
// through LDS via global_load_async_to_lds_b128 (ASYNCcnt pipeline).
// The S4 convolution uses the exact diagonal recurrence (3e10 FLOP) instead
// of an O(L^2) conv (2.7e14 FLOP) or a length-8194 FFT.
// ---------------------------------------------------------------------------

#define Hdim   256
#define Nst    64
#define NSPIN  4096
#define Bsz    64
#define Lseq   (NSPIN + 1)          // 4097
#define Mrows  (Lseq * Bsz)         // 262208  (multiple of 16)
#define MT     (Mrows / 16)         // 16388 m-tiles (even)
#define KSTEPS (Hdim / 32)          // 8 WMMA k-steps
#define N2     (2 * Hdim)           // 512 (GLU pre-activation width)
#define PI_F   3.14159265358979f

typedef __bf16 bf16_t;
typedef __attribute__((ext_vector_type(16))) __bf16 v16bf;
typedef __attribute__((ext_vector_type(8)))  float  v8f;

// --- WMMA fragment layouts (CDNA5 ISA §7.12.2, wave32) -----------
// A (16x32 bf16): lanes 0-15 hold K 0-7 (v0-3) & 16-23 (v4-7); lanes 16-31
// hold K 8-15 & 24-31; each VGPR packs 2 consecutive K in [15:0],[31:16].
__device__ __forceinline__ size_t apack_idx(int m, int k) {
  int mtile = m >> 4, mrow = m & 15;
  int kstep = k >> 5, kk = k & 31;
  int group = kk >> 3;                          // 0..3
  int lane  = mrow + ((group & 1) << 4);
  int vg    = ((kk & 7) >> 1) + ((group >> 1) << 2);
  int slot  = (vg << 1) + (kk & 1);
  return ((((size_t)mtile * KSTEPS + kstep) * 32 + lane) << 4) + slot;
}
// B (32x16 bf16, KxN): same K order as A: lanes 0-15 hold K 0-7 & 16-23;
// lanes 16-31 hold K 8-15 & 24-31 (N=lane&15).
__device__ __forceinline__ size_t bpack_idx(int k, int n) {
  int ntile = n >> 4, nrow = n & 15;
  int kstep = k >> 5, kk = k & 31;
  int lane  = nrow + (((kk >> 3) & 1) << 4);
  int vg    = ((kk & 7) >> 1) + ((kk >> 4) << 2);
  int slot  = (vg << 1) + (kk & 1);
  return ((((size_t)ntile * KSTEPS + kstep) * 32 + lane) << 4) + slot;
}

__device__ __forceinline__ float wave_red32(float v) {
  #pragma unroll
  for (int m = 16; m >= 1; m >>= 1) v += __shfl_xor(v, m, 32);
  return v;
}

__device__ __forceinline__ float gelu_tanh(float y) {
  return 0.5f * y * (1.0f + tanhf(0.7978845608f * (y + 0.044715f * y * y * y)));
}

// --- Kernel 1: S4D ZOH-style discretization: lambda = exp(dt*A), Ct ---------
__global__ __launch_bounds__(256) void k_params(
    const float* __restrict__ log_dt, const float* __restrict__ A_re,
    const float* __restrict__ A_im,  const float* __restrict__ C_re,
    const float* __restrict__ C_im,
    float* __restrict__ lam_r, float* __restrict__ lam_i,
    float* __restrict__ ct_r,  float* __restrict__ ct_i) {
  int i = blockIdx.x * blockDim.x + threadIdx.x;          // H*N threads
  int h = i / Nst;
  float dt  = expf(log_dt[h]);
  float are = -expf(A_re[i]);                             // Re(A)
  float aim = A_im[i];                                    // Im(A)
  float dr = dt * are, di = dt * aim;
  float er = expf(dr);
  float lr = er * cosf(di), li = er * sinf(di);           // lambda = exp(dt*A)
  // q = (lambda - 1) / A ;  Ct = C * q
  float nr = lr - 1.0f, ni = li;
  float den = are * are + aim * aim;
  float qr = (nr * are + ni * aim) / den;
  float qi = (ni * are - nr * aim) / den;
  float cr = C_re[i], ci = C_im[i];
  lam_r[i] = lr; lam_i[i] = li;
  ct_r[i] = cr * qr - ci * qi;
  ct_i[i] = cr * qi + ci * qr;
}

// --- Kernel 2: pre-linear; one-hot spins select a single W_pre row ----------
__global__ __launch_bounds__(256) void k_pre(
    const int* __restrict__ spins, const float* __restrict__ salt,
    const float* __restrict__ W_pre, const float* __restrict__ b_pre,
    float* __restrict__ x) {
  size_t idx = (size_t)blockIdx.x * blockDim.x + threadIdx.x;  // L*B*H threads
  int h = (int)(idx % Hdim);
  size_t lb = idx / Hdim;
  int b = (int)(lb % Bsz);
  int l = (int)(lb / Bsz);
  float acc = b_pre[h];
  #pragma unroll
  for (int j = 0; j < 5; ++j) acc += salt[j] * W_pre[(2 + j) * Hdim + h];
  if (l > 0) {
    int s = spins[(size_t)(l - 1) * Bsz + b];
    acc += W_pre[s * Hdim + h];
  }
  x[idx] = acc;                                            // (L,B,H) row-major
}

// --- Kernel 3: pack W_glu (256,512) f32 -> bf16 B-fragment layout -----------
__global__ __launch_bounds__(256) void k_bpack(
    const float* __restrict__ W_glu, bf16_t* __restrict__ Bpack) {
  int i = blockIdx.x * blockDim.x + threadIdx.x;           // 256*512 threads
  int k = i / N2, n = i % N2;
  Bpack[bpack_idx(k, n)] = (bf16_t)W_glu[(size_t)k * N2 + n];
}

// --- Kernel 4: diagonal S4 recurrence, one wave per (b,h), 2 states/lane ----
//   s_l = lambda*s_{l-1} + u_l ; y = 2*Re(Ct.s) + D*u ; A_pack = bf16(gelu(y))
__global__ __launch_bounds__(256) void k_scan(
    const float* __restrict__ x, const float* __restrict__ lam_r,
    const float* __restrict__ lam_i, const float* __restrict__ ct_r,
    const float* __restrict__ ct_i, const float* __restrict__ Dv,
    bf16_t* __restrict__ Apack) {
  int gid  = blockIdx.x * blockDim.x + threadIdx.x;
  int w    = gid >> 5;                                     // 0..B*H-1
  int lane = gid & 31;
  int b = w >> 8;                                          // w / H
  int h = w & (Hdim - 1);
  int i0 = h * Nst + lane, i1 = i0 + 32;
  float l0r = lam_r[i0], l0i = lam_i[i0], c0r = ct_r[i0], c0i = ct_i[i0];
  float l1r = lam_r[i1], l1i = lam_i[i1], c1r = ct_r[i1], c1i = ct_i[i1];
  float Dh = Dv[h];
  float s0r = 0.f, s0i = 0.f, s1r = 0.f, s1i = 0.f;
  const float* up = x + (size_t)b * Hdim + h;              // stride B*H over l
  for (int l = 0; l < Lseq; ++l) {
    float u = up[(size_t)l * (Bsz * Hdim)];                // wave-uniform load
    float t0r = fmaf(l0r, s0r, fmaf(-l0i, s0i, u));
    float t0i = fmaf(l0r, s0i,  l0i * s0r);
    float t1r = fmaf(l1r, s1r, fmaf(-l1i, s1i, u));
    float t1i = fmaf(l1r, s1i,  l1i * s1r);
    s0r = t0r; s0i = t0i; s1r = t1r; s1i = t1i;
    float p = fmaf(c0r, s0r, -c0i * s0i) + fmaf(c1r, s1r, -c1i * s1i);
    p = wave_red32(p);
    if (lane == 0) {
      float y = 2.0f * p + Dh * u;
      Apack[apack_idx(l * Bsz + b, h)] = (bf16_t)gelu_tanh(y);
    }
  }
}

// --- Kernel 5: GLU linear on WMMA -------------------------------------------
// Block = 8 waves = 2 m-tiles x 4 column-slices. Wave (mtile, t) owns value
// n-tiles {t,t+4,t+8,t+12} and their GLU gate tiles (+16) -> 8 accumulators,
// so every A fragment is reused by 8 WMMAs. W_glu B-fragments are double-
// buffered through LDS with global_load_async_to_lds_b128 (32 KB per k-step,
// 64 KB total), staged one k-step ahead of the WMMA consumption.
__global__ __launch_bounds__(256) void k_gemm_glu(
    const bf16_t* __restrict__ Apack, const bf16_t* __restrict__ Bpack,
    const float* __restrict__ b_glu, bf16_t* __restrict__ x2) {
  __shared__ __align__(32) bf16_t ldsB[2 * 32 * 32 * 16];  // 64 KB, 2 buffers
  const int tid   = threadIdx.x;
  const int wloc  = tid >> 5;
  const int lane  = tid & 31;
  const int mtile = blockIdx.x * 2 + (wloc >> 2);
  const int t     = wloc & 3;

  const unsigned long long bbase = (unsigned long long)(size_t)Bpack;

  // Stage all 32 n-tiles of k-step `ks` into LDS buffer `buf` (async DMA).
  auto stage = [&](int ks, int buf) {
    #pragma unroll
    for (int i = 0; i < 4; ++i) {
      int unit = tid + 256 * i;                        // 0..1023 = (nt, ln)
      int nt = unit >> 5, ln = unit & 31;
      unsigned long long g =
          bbase + (unsigned long long)(((nt * KSTEPS + ks) * 32 + ln) * 32);
      unsigned l = (unsigned)(size_t)&ldsB[(size_t)buf * 16384 +
                                           (nt * 32 + ln) * 16];
      asm volatile("global_load_async_to_lds_b128 %0, %1, off"
                   :: "v"(l), "v"(g) : "memory");
      asm volatile("global_load_async_to_lds_b128 %0, %1, off"
                   :: "v"(l + 16u), "v"(g + 16ull) : "memory");
    }
  };

  const v16bf* A = (const v16bf*)Apack;
  size_t abase = ((size_t)mtile * KSTEPS) * 32 + lane;

  v8f acc[8];
  #pragma unroll
  for (int j = 0; j < 8; ++j) acc[j] = (v8f){};

  stage(0, 0);
  asm volatile("s_wait_asynccnt 0x0" ::: "memory");
  __syncthreads();

  #pragma unroll
  for (int ks = 0; ks < KSTEPS; ++ks) {
    if (ks + 1 < KSTEPS) stage(ks + 1, (ks + 1) & 1);
    v16bf a = A[abase + (size_t)ks * 32];
    if (ks + 1 < KSTEPS)
      __builtin_prefetch((const void*)&A[abase + (size_t)(ks + 1) * 32], 0, 0);
    #pragma unroll
    for (int j = 0; j < 8; ++j) {
      int nt = (j < 4) ? (t + 4 * j) : (16 + t + 4 * (j - 4));
      v16bf bf = *(const v16bf*)&ldsB[(size_t)(ks & 1) * 16384 +
                                      (nt * 32 + lane) * 16];
      acc[j] = __builtin_amdgcn_wmma_f32_16x16x32_bf16(
          false, a, false, bf, (short)0, acc[j], false, false);
    }
    asm volatile("s_wait_asynccnt 0x0" ::: "memory");
    __syncthreads();
  }

  // C/D layout: lanes 0-15 rows v, lanes 16-31 rows v+8; N = lane&15.
  int nl  = lane & 15;
  int mhi = (lane >> 4) << 3;
  #pragma unroll
  for (int j = 0; j < 4; ++j) {
    int n = (t + 4 * j) * 16 + nl;
    #pragma unroll
    for (int v = 0; v < 8; ++v) {
      int m = mtile * 16 + v + mhi;
      float z0 = acc[j][v]     + b_glu[n];
      float z1 = acc[j + 4][v] + b_glu[n + Hdim];
      float g  = z0 / (1.0f + expf(-z1));                  // z0 * sigmoid(z1)
      x2[(size_t)m * Hdim + n] = (bf16_t)g;
    }
  }
}

// --- Kernel 6: amp/phase heads, one wave per row m = l*B+b ------------------
__global__ __launch_bounds__(256) void k_heads(
    const bf16_t* __restrict__ x2,
    const float* __restrict__ W_amp, const float* __restrict__ b_amp,
    const float* __restrict__ W_ph,  const float* __restrict__ b_ph,
    float* __restrict__ out) {
  int gid  = blockIdx.x * blockDim.x + threadIdx.x;
  int m    = gid >> 5;
  int lane = gid & 31;
  float a0 = 0.f, a1 = 0.f, p0 = 0.f, p1 = 0.f;
  const bf16_t* row = x2 + (size_t)m * Hdim;
  #pragma unroll
  for (int j = 0; j < 8; ++j) {
    int h = lane + 32 * j;
    float v = (float)row[h];
    a0 = fmaf(v, W_amp[h * 2 + 0], a0);
    a1 = fmaf(v, W_amp[h * 2 + 1], a1);
    p0 = fmaf(v, W_ph[h * 2 + 0], p0);
    p1 = fmaf(v, W_ph[h * 2 + 1], p1);
  }
  a0 = wave_red32(a0); a1 = wave_red32(a1);
  p0 = wave_red32(p0); p1 = wave_red32(p1);
  if (lane == 0) {
    a0 += b_amp[0]; a1 += b_amp[1];
    float mx  = fmaxf(a0, a1);
    float lse = mx + logf(expf(a0 - mx) + expf(a1 - mx));
    out[(size_t)m * 2 + 0] = a0 - lse;
    out[(size_t)m * 2 + 1] = a1 - lse;
    float t0 = p0 + b_ph[0], t1 = p1 + b_ph[1];
    size_t po = (size_t)Mrows * 2;
    out[po + (size_t)m * 2 + 0] = 2.0f * PI_F * (1.0f + t0 / (1.0f + fabsf(t0)));
    out[po + (size_t)m * 2 + 1] = 2.0f * PI_F * (1.0f + t1 / (1.0f + fabsf(t1)));
  }
}

extern "C" void kernel_launch(void* const* d_in, const int* in_sizes, int n_in,
                              void* d_out, int out_size, void* d_ws, size_t ws_size,
                              hipStream_t stream) {
  const int*   spins  = (const int*)  d_in[0];
  const float* salt   = (const float*)d_in[1];
  const float* W_pre  = (const float*)d_in[2];
  const float* b_pre  = (const float*)d_in[3];
  const float* log_dt = (const float*)d_in[4];
  const float* A_re   = (const float*)d_in[5];
  const float* A_im   = (const float*)d_in[6];
  const float* C_re   = (const float*)d_in[7];
  const float* C_im   = (const float*)d_in[8];
  const float* Dv     = (const float*)d_in[9];
  const float* W_glu  = (const float*)d_in[10];
  const float* b_glu  = (const float*)d_in[11];
  const float* W_amp  = (const float*)d_in[12];
  const float* b_amp  = (const float*)d_in[13];
  const float* W_ph   = (const float*)d_in[14];
  const float* b_ph   = (const float*)d_in[15];
  float* out = (float*)d_out;

  // Workspace partition (256B aligned slabs).
  size_t off = 0;
  auto take = [&](size_t bytes) {
    void* p = (char*)d_ws + off;
    off += (bytes + 255) & ~(size_t)255;
    return p;
  };
  float*  lam_r = (float*)take((size_t)Hdim * Nst * 4);
  float*  lam_i = (float*)take((size_t)Hdim * Nst * 4);
  float*  ct_r  = (float*)take((size_t)Hdim * Nst * 4);
  float*  ct_i  = (float*)take((size_t)Hdim * Nst * 4);
  float*  x     = (float*)take((size_t)Mrows * Hdim * 4);     // pre-linear out
  bf16_t* Apack = (bf16_t*)take((size_t)Mrows * Hdim * 2);    // gelu(y) frags
  bf16_t* Bpack = (bf16_t*)take((size_t)Hdim * N2 * 2);       // W_glu frags
  bf16_t* x2    = (bf16_t*)take((size_t)Mrows * Hdim * 2);    // GLU output
  (void)ws_size; (void)n_in; (void)in_sizes; (void)out_size;

  k_params<<<(Hdim * Nst) / 256, 256, 0, stream>>>(log_dt, A_re, A_im, C_re,
                                                   C_im, lam_r, lam_i, ct_r, ct_i);
  k_pre<<<(unsigned)(((size_t)Mrows * Hdim) / 256), 256, 0, stream>>>(
      spins, salt, W_pre, b_pre, x);
  k_bpack<<<(Hdim * N2) / 256, 256, 0, stream>>>(W_glu, Bpack);
  k_scan<<<(Bsz * Hdim * 32) / 256, 256, 0, stream>>>(x, lam_r, lam_i, ct_r,
                                                      ct_i, Dv, Apack);
  k_gemm_glu<<<MT / 2, 256, 0, stream>>>(Apack, Bpack, b_glu, x2);
  k_heads<<<(Mrows * 32) / 256, 256, 0, stream>>>(x2, W_amp, b_amp, W_ph, b_ph, out);
}
